// SocialAttention_62088047231245
// MI455X (gfx1250) — matrix operand order repaired and checked by reading, not compiled, not run on hardware
//
#include <hip/hip_runtime.h>
#include <hip/hip_bf16.h>
#include <math.h>

#define H      128
#define NFREQ  64
#define AA     64
#define MM     6
#define TT     50
#define NN     (AA*MM*TT)     // 19200
#define EE     250000
#define KP     160            // padded K stride in LDS (covers K=129)
#define SLOT   (H*KP)         // elements per staged weight image
#define PI_F   3.14159265358979f

typedef __attribute__((ext_vector_type(16))) __bf16 v16bf;
typedef __attribute__((ext_vector_type(8)))  __bf16 v8bf;
typedef __attribute__((ext_vector_type(8)))  float  v8f;

// ---------------- WMMA helpers ----------------

static __device__ inline v8f wmma_bf16(v16bf a, v16bf b, v8f c) {
  // D = A(16x32 bf16) x B(32x16 bf16) + C(16x16 f32)
  return __builtin_amdgcn_wmma_f32_16x16x32_bf16(false, a, false, b, (short)0, c, false, false);
}

static __device__ inline void zero8(v8f acc[8]) {
  v8f z = {0.f,0.f,0.f,0.f,0.f,0.f,0.f,0.f};
#pragma unroll
  for (int j = 0; j < 8; ++j) acc[j] = z;
}

// A fragment: 16-bit A 16x32, lane (half,mrow): row=mrow,
// e=0..7 -> K = kb + half*8 + e ; e=8..15 -> K = kb + 16 + half*8 + (e-8)
static __device__ inline v16bf frag_a(const __bf16* rowp, int kb, int half) {
  const __bf16* p = rowp + kb + half*8;
  v8bf lo = *(const v8bf*)(p);
  v8bf hi = *(const v8bf*)(p + 16);
  v16bf a;
#pragma unroll
  for (int e = 0; e < 8; ++e) { a[e] = lo[e]; a[e+8] = hi[e]; }
  return a;
}

static __device__ inline v16bf frag_zero() {
  v16bf a;
#pragma unroll
  for (int e = 0; e < 16; ++e) a[e] = (__bf16)0.f;
  return a;
}

// B fragment from column-major LDS weights wcm[col*KP + k]:
// lane (half,mrow): col = given, e=0..7 -> K = kb + half*8 + e ; e=8..15 -> K = kb + 16 + half*8 + (e-8)
static __device__ inline v16bf frag_b(const __bf16* wcm, int col, int kb, int half) {
  const __bf16* p = wcm + col*KP + kb + half*8;
  v8bf lo = *(const v8bf*)(p);
  v8bf hi = *(const v8bf*)(p + 16);
  v16bf b;
#pragma unroll
  for (int e = 0; e < 8; ++e) { b[e] = lo[e]; b[e+8] = hi[e]; }
  return b;
}

// Async-copy a pre-converted bf16 weight image (H*KP elements, already in the
// exact LDS layout) from global scratch into LDS using the CDNA5 async path:
// GLOBAL_LOAD_ASYNC_TO_LDS_B128 (tracked on ASYNCcnt, no VGPR round-trip).
static __device__ inline void stage_async(const __bf16* __restrict__ src, __bf16* dst) {
  unsigned long long sbase = (unsigned long long)(uintptr_t)src;
  for (int idx = threadIdx.x * 8; idx < SLOT; idx += blockDim.x * 8) {
    unsigned lds = (unsigned)(uintptr_t)(dst + idx);   // LDS byte offset (low 32 bits)
    unsigned off = (unsigned)(idx * sizeof(__bf16));   // global byte offset
    asm volatile("global_load_async_to_lds_b128 %0, %1, %2"
                 :: "v"(lds), "v"(off), "s"(sbase)
                 : "memory");
  }
  asm volatile("s_wait_asynccnt 0x0" ::: "memory");
}

// 16xH GEMM tile: acc[j] += A(16xK=128) * W(:, j*16+mrow block)
static __device__ inline void gemm_acc(const __bf16* A, int Astride, int valid_rows,
                                       const __bf16* wcm, int mrow, int half, v8f acc[8]) {
  for (int kb = 0; kb < H; kb += 32) {
    v16bf a = (mrow < valid_rows) ? frag_a(A + (long)mrow*Astride, kb, half) : frag_zero();
#pragma unroll
    for (int j = 0; j < 8; ++j) {
      v16bf b = frag_b(wcm, j*16 + mrow, kb, half);
      acc[j] = wmma_bf16(a, b, acc[j]);
    }
  }
}

// ---------------- reductions / LN ----------------

static __device__ inline float redsum16(float v) {
  v += __shfl_xor(v, 1, 32);
  v += __shfl_xor(v, 2, 32);
  v += __shfl_xor(v, 4, 32);
  v += __shfl_xor(v, 8, 32);
  return v;
}

// Row-wise LayerNorm over 128 cols held in 8 WMMA D tiles.
static __device__ inline void ln_rows(v8f acc[8], int mrow, const float* g, const float* b,
                                      bool do_relu) {
  float gv[8], bv[8];
#pragma unroll
  for (int j = 0; j < 8; ++j) { gv[j] = g[j*16+mrow]; bv[j] = b[j*16+mrow]; }
#pragma unroll
  for (int r = 0; r < 8; ++r) {
    float s = 0.f;
#pragma unroll
    for (int j = 0; j < 8; ++j) s += acc[j][r];
    float mu = redsum16(s) * (1.f/H);
    float vs = 0.f;
#pragma unroll
    for (int j = 0; j < 8; ++j) { float d = acc[j][r] - mu; vs += d*d; }
    float inv = rsqrtf(redsum16(vs) * (1.f/H) + 1e-5f);
#pragma unroll
    for (int j = 0; j < 8; ++j) {
      float y = (acc[j][r] - mu) * inv * gv[j] + bv[j];
      acc[j][r] = do_relu ? fmaxf(y, 0.f) : y;
    }
  }
}

static __device__ inline unsigned fkey(float f) {
  unsigned u = __float_as_uint(f);
  return (u & 0x80000000u) ? ~u : (u | 0x80000000u);
}
static __device__ inline float unfkey(unsigned u) {
  return (u & 0x80000000u) ? __uint_as_float(u & 0x7fffffffu) : __uint_as_float(~u);
}

// ---------------- kernels ----------------

// One-time: convert [K x 128] row-major f32 weight into the exact LDS image:
// column-major bf16, stride KP, zero-padded past K.
__global__ void k_conv(const float* __restrict__ W, int ldw, int K, __bf16* __restrict__ dst) {
  int idx = blockIdx.x * blockDim.x + threadIdx.x;
  if (idx >= SLOT) return;
  int n = idx / KP, k = idx - n*KP;
  dst[idx] = (__bf16)((k < K) ? W[(long)k*ldw + n] : 0.f);
}

__global__ void k_edge_geom(const int* __restrict__ edges, const float* __restrict__ pos,
                            const float* __restrict__ head, float* __restrict__ rel) {
  int e = blockIdx.x * blockDim.x + threadIdx.x;
  if (e >= EE) return;
  int s = edges[e], d = edges[EE + e];
  int as = s & 63, ms = (s >> 6) % MM, ts = s / (AA*MM);
  int ad = d & 63, md = (d >> 6) % MM, td = d / (AA*MM);
  long ps = (long)(as*MM + ms)*TT + ts;
  long pd = (long)(ad*MM + md)*TT + td;
  float relx = pos[ps*2+0] - pos[pd*2+0];
  float rely = pos[ps*2+1] - pos[pd*2+1];
  float hs = head[ps], hd = head[pd];
  float dist = sqrtf(relx*relx + rely*rely);
  float cx = cosf(hd), cy = sinf(hd);
  float dir = atan2f(cx*rely - cy*relx, cx*relx + cy*rely);
  float rh = fmodf(hs - hd + PI_F, 2.f*PI_F);
  if (rh < 0.f) rh += 2.f*PI_F;
  rh -= PI_F;
  rel[e*3+0] = dist; rel[e*3+1] = dir; rel[e*3+2] = rh;
}

// warp-per-node: permute x -> xf, LN -> xn (bf16)
__global__ void k_node_prep(const float* __restrict__ x, const float* __restrict__ g,
                            const float* __restrict__ b, float* __restrict__ xf,
                            __bf16* __restrict__ xn) {
  int warp = (blockIdx.x * blockDim.x + threadIdx.x) >> 5;
  int lane = threadIdx.x & 31;
  if (warp >= NN) return;
  int n = warp;
  int a = n & 63, m = (n >> 6) % MM, t = n / (AA*MM);
  const float* src = x + ((long)(a*MM + m)*TT + t) * H;
  float v[4];
#pragma unroll
  for (int i = 0; i < 4; ++i) v[i] = src[lane + i*32];
  float s = v[0] + v[1] + v[2] + v[3];
#pragma unroll
  for (int msk = 1; msk < 32; msk <<= 1) s += __shfl_xor(s, msk, 32);
  float mu = s * (1.f/H);
  float vs = 0.f;
#pragma unroll
  for (int i = 0; i < 4; ++i) { float d = v[i] - mu; vs += d*d; }
#pragma unroll
  for (int msk = 1; msk < 32; msk <<= 1) vs += __shfl_xor(vs, msk, 32);
  float inv = rsqrtf(vs * (1.f/H) + 1e-5f);
#pragma unroll
  for (int i = 0; i < 4; ++i) {
    int c = lane + i*32;
    xf[(long)n*H + c] = v[i];
    xn[(long)n*H + c] = (__bf16)((v[i] - mu) * inv * g[c] + b[c]);
  }
}

// q/k/v projections: 4 waves x 16 nodes per block. Weight slots 0,1,2.
__global__ void k_qkv(const __bf16* __restrict__ xn, const __bf16* __restrict__ wsl,
                      const float* bq, const float* bv,
                      float* __restrict__ q, float* __restrict__ k, float* __restrict__ v) {
  __shared__ __attribute__((aligned(16))) __bf16 wcm[SLOT];
  int wave = threadIdx.x >> 5, lane = threadIdx.x & 31;
  int mrow = lane & 15, half = lane >> 4;
  long row0 = (long)blockIdx.x*64 + wave*16;
  const __bf16* A = xn + row0*H;

  stage_async(wsl + 0*SLOT, wcm); __syncthreads();
  { v8f acc[8]; zero8(acc); gemm_acc(A, H, 16, wcm, mrow, half, acc);
#pragma unroll
    for (int j = 0; j < 8; ++j) { int col = j*16+mrow; float bb = bq[col];
#pragma unroll
      for (int r = 0; r < 8; ++r) q[(row0 + r + half*8)*H + col] = acc[j][r] + bb; } }
  __syncthreads();
  stage_async(wsl + 1*SLOT, wcm); __syncthreads();
  { v8f acc[8]; zero8(acc); gemm_acc(A, H, 16, wcm, mrow, half, acc);
#pragma unroll
    for (int j = 0; j < 8; ++j) { int col = j*16+mrow;
#pragma unroll
      for (int r = 0; r < 8; ++r) k[(row0 + r + half*8)*H + col] = acc[j][r]; } }
  __syncthreads();
  stage_async(wsl + 2*SLOT, wcm); __syncthreads();
  { v8f acc[8]; zero8(acc); gemm_acc(A, H, 16, wcm, mrow, half, acc);
#pragma unroll
    for (int j = 0; j < 8; ++j) { int col = j*16+mrow; float bb = bv[col];
#pragma unroll
      for (int r = 0; r < 8; ++r) v[(row0 + r + half*8)*H + col] = acc[j][r] + bb; } }
}

// Fourier embedding + 3 MLPs + out proj + pre_r LN. Slots 3..9. 4 waves x 16 edges.
__global__ void k_fourier(const float* __restrict__ rel, const float* __restrict__ freqs,
                          const __bf16* __restrict__ wsl,
                          const float* mlp_b1, const float* mlp_ln_g, const float* mlp_ln_b,
                          const float* mlp_b2, const float* out_ln_g, const float* out_ln_b,
                          const float* out_b, const float* pre_r_g, const float* pre_r_b,
                          __bf16* __restrict__ rn) {
  __shared__ __attribute__((aligned(16))) __bf16 wcm[SLOT];
  __shared__ __attribute__((aligned(16))) __bf16 hbuf[4*16*H];
  __shared__ float sfreq[3*NFREQ];
  int wave = threadIdx.x >> 5, lane = threadIdx.x & 31;
  int mrow = lane & 15, half = lane >> 4;
  long tile0 = (long)blockIdx.x*64 + wave*16;
  __bf16* hb = hbuf + wave*16*H;
  for (int i = threadIdx.x; i < 3*NFREQ; i += blockDim.x) sfreq[i] = freqs[i] * (2.f*PI_F);

  v8f accT[8]; zero8(accT);
  long grow_a = tile0 + mrow;

  for (int i = 0; i < 3; ++i) {
    __syncthreads();
    stage_async(wsl + (3 + i)*SLOT, wcm);   // mlp_w1[i], K=129 (padded image)
    __syncthreads();
    float reli = (grow_a < EE) ? rel[grow_a*3 + i] : 0.f;
    v8f acc1[8]; zero8(acc1);
    for (int kb = 0; kb < KP; kb += 32) {
      v16bf a;
#pragma unroll
      for (int e2 = 0; e2 < 16; ++e2) {
        int vv = e2 >> 1, ss = e2 & 1;
        int kk = kb + ((vv >= 4) ? 16 : 0) + ((vv & 3)*2 + ss) + half*8;
        float val;
        if (kk < NFREQ)             val = __cosf(reli * sfreq[i*NFREQ + kk]);
        else if (kk < 2*NFREQ)      val = __sinf(reli * sfreq[i*NFREQ + kk - NFREQ]);
        else if (kk == 2*NFREQ)     val = reli;
        else                        val = 0.f;
        a[e2] = (__bf16)val;
      }
#pragma unroll
      for (int j = 0; j < 8; ++j) {
        v16bf b = frag_b(wcm, j*16 + mrow, kb, half);
        acc1[j] = wmma_bf16(a, b, acc1[j]);
      }
    }
#pragma unroll
    for (int j = 0; j < 8; ++j) { float bb = mlp_b1[i*H + j*16 + mrow];
#pragma unroll
      for (int r = 0; r < 8; ++r) acc1[j][r] += bb; }
    ln_rows(acc1, mrow, mlp_ln_g + i*H, mlp_ln_b + i*H, true);
    __syncthreads();
#pragma unroll
    for (int j = 0; j < 8; ++j)
#pragma unroll
      for (int r = 0; r < 8; ++r)
        hb[(r + half*8)*H + j*16 + mrow] = (__bf16)acc1[j][r];
    stage_async(wsl + (6 + i)*SLOT, wcm);   // mlp_w2[i]
    __syncthreads();
    gemm_acc(hb, H, 16, wcm, mrow, half, accT);
#pragma unroll
    for (int j = 0; j < 8; ++j) { float bb = mlp_b2[i*H + j*16 + mrow];
#pragma unroll
      for (int r = 0; r < 8; ++r) accT[j][r] += bb; }
  }

  ln_rows(accT, mrow, out_ln_g, out_ln_b, true);
  __syncthreads();
#pragma unroll
  for (int j = 0; j < 8; ++j)
#pragma unroll
    for (int r = 0; r < 8; ++r)
      hb[(r + half*8)*H + j*16 + mrow] = (__bf16)accT[j][r];
  stage_async(wsl + 9*SLOT, wcm);           // out_w
  __syncthreads();
  v8f accR[8]; zero8(accR);
  gemm_acc(hb, H, 16, wcm, mrow, half, accR);
#pragma unroll
  for (int j = 0; j < 8; ++j) { float bb = out_b[j*16 + mrow];
#pragma unroll
    for (int r = 0; r < 8; ++r) accR[j][r] += bb; }
  ln_rows(accR, mrow, pre_r_g, pre_r_b, false);
#pragma unroll
  for (int j = 0; j < 8; ++j) { int col = j*16 + mrow;
#pragma unroll
    for (int r = 0; r < 8; ++r) {
      long row = tile0 + r + half*8;
      if (row < EE) rn[row*H + col] = (__bf16)accR[j][r];
    } }
}

// kr/vr projections, ke=k[src]+kr, sim + atomicMax, ve store. Slots 10,11.
__global__ void k_attn(const __bf16* __restrict__ rn, const int* __restrict__ edges,
                       const __bf16* __restrict__ wsl, const float* bvr,
                       const float* __restrict__ qb, const float* __restrict__ kbuf,
                       const float* __restrict__ vbuf,
                       float* __restrict__ sim, unsigned* __restrict__ smaxu,
                       __bf16* __restrict__ ve) {
  __shared__ __attribute__((aligned(16))) __bf16 wcm[SLOT];
  int wave = threadIdx.x >> 5, lane = threadIdx.x & 31;
  int mrow = lane & 15, half = lane >> 4;
  long tile0 = (long)blockIdx.x*64 + wave*16;
  long rem = (long)EE - tile0;
  int valid = rem >= 16 ? 16 : (rem > 0 ? (int)rem : 0);
  const __bf16* A = rn + tile0*H;

  stage_async(wsl + 10*SLOT, wcm); __syncthreads();
  v8f accK[8]; zero8(accK);
  gemm_acc(A, H, valid, wcm, mrow, half, accK);
#pragma unroll
  for (int j = 0; j < 8; ++j) {
    int col = j*16 + mrow;
#pragma unroll
    for (int r = 0; r < 8; ++r) {
      long row = tile0 + r + half*8;
      float part = 0.f; int d = 0;
      if (row < EE) {
        int s = edges[row]; d = edges[EE + row];
        float ke = accK[j][r] + kbuf[(long)s*H + col];
        part = ke * qb[(long)d*H + col];
      }
      float sv = redsum16(part) * 0.25f;            // HEAD_DIM^-0.5
      if (row < EE && mrow == 0) {
        sim[row*8 + j] = sv;
        atomicMax(smaxu + (long)d*8 + j, fkey(sv));
      }
    }
  }
  __syncthreads();
  stage_async(wsl + 11*SLOT, wcm); __syncthreads();
  v8f accV[8]; zero8(accV);
  gemm_acc(A, H, valid, wcm, mrow, half, accV);
#pragma unroll
  for (int j = 0; j < 8; ++j) {
    int col = j*16 + mrow; float bb = bvr[col];
#pragma unroll
    for (int r = 0; r < 8; ++r) {
      long row = tile0 + r + half*8;
      if (row < EE) {
        int s = edges[row];
        ve[row*H + col] = (__bf16)(accV[j][r] + bb + vbuf[(long)s*H + col]);
      }
    }
  }
}

__global__ void k_soft2(const int* __restrict__ edges, const unsigned* __restrict__ smaxu,
                        float* __restrict__ sim, float* __restrict__ den) {
  long t = (long)blockIdx.x * blockDim.x + threadIdx.x;
  if (t >= (long)EE*8) return;
  long e = t >> 3; int h = (int)(t & 7);
  int d = edges[EE + e];
  float ex = __expf(sim[t] - unfkey(smaxu[(long)d*8 + h]));
  sim[t] = ex;
  atomicAdd(den + (long)d*8 + h, ex);
}

__global__ void k_agg(const int* __restrict__ edges, const float* __restrict__ sim,
                      const float* __restrict__ den, const __bf16* __restrict__ ve,
                      float* __restrict__ agg) {
  long t = (long)blockIdx.x * blockDim.x + threadIdx.x;
  if (t >= (long)EE*H) return;
  long e = t >> 7; int c = (int)(t & 127); int h = c >> 4;
  int d = edges[EE + e];
  float w = sim[e*8 + h] / den[(long)d*8 + h];
  atomicAdd(agg + (long)d*H + c, (float)ve[t] * w);
}

// gate/update/out-projection. Slots 12 (wg lo), 13 (wg hi), 14 (ws), 15 (wo).
__global__ void k_update(const __bf16* __restrict__ xn, const float* __restrict__ xf,
                         const float* __restrict__ agg, const __bf16* __restrict__ wsl,
                         const float* bg, const float* bs, const float* bo,
                         float* __restrict__ xmid) {
  __shared__ __attribute__((aligned(16))) __bf16 wcm[SLOT];
  __shared__ __attribute__((aligned(16))) __bf16 hbuf[4*16*H];
  int wave = threadIdx.x >> 5, lane = threadIdx.x & 31;
  int mrow = lane & 15, half = lane >> 4;
  long row0 = (long)blockIdx.x*64 + wave*16;
  __bf16* hb = hbuf + wave*16*H;

  // gate: [agg, xn] @ wg  — first half K uses agg (f32 -> bf16 on the fly)
  stage_async(wsl + 12*SLOT, wcm); __syncthreads();
  v8f accG[8]; zero8(accG);
  for (int kb = 0; kb < H; kb += 32) {
    v16bf a;
#pragma unroll
    for (int e2 = 0; e2 < 16; ++e2) {
      int vv = e2 >> 1, ss = e2 & 1;
      int kk = kb + ((vv >= 4) ? 16 : 0) + ((vv & 3)*2 + ss) + half*8;
      a[e2] = (__bf16)agg[(row0 + mrow)*H + kk];
    }
#pragma unroll
    for (int j = 0; j < 8; ++j)
      accG[j] = wmma_bf16(a, frag_b(wcm, j*16 + mrow, kb, half), accG[j]);
  }
  __syncthreads();
  stage_async(wsl + 13*SLOT, wcm); __syncthreads();
  gemm_acc(xn + row0*H, H, 16, wcm, mrow, half, accG);
  __syncthreads();
  stage_async(wsl + 14*SLOT, wcm); __syncthreads();
  v8f accS[8]; zero8(accS);
  gemm_acc(xn + row0*H, H, 16, wcm, mrow, half, accS);

#pragma unroll
  for (int j = 0; j < 8; ++j) {
    int col = j*16 + mrow; float bgv = bg[col], bsv = bs[col];
#pragma unroll
    for (int r = 0; r < 8; ++r) {
      long row = row0 + r + half*8;
      float av = agg[row*H + col];
      float gv = 1.f / (1.f + __expf(-(accG[j][r] + bgv)));
      float sv = accS[j][r] + bsv;
      hb[(r + half*8)*H + col] = (__bf16)(av + gv * (sv - av));
    }
  }
  __syncthreads();
  stage_async(wsl + 15*SLOT, wcm); __syncthreads();
  v8f accO[8]; zero8(accO);
  gemm_acc(hb, H, 16, wcm, mrow, half, accO);
#pragma unroll
  for (int j = 0; j < 8; ++j) {
    int col = j*16 + mrow; float bov = bo[col];
#pragma unroll
    for (int r = 0; r < 8; ++r) {
      long row = row0 + r + half*8;
      xmid[row*H + col] = xf[row*H + col] + accO[j][r] + bov;
    }
  }
}

__global__ void k_ffnprep(const float* __restrict__ xmid, const float* __restrict__ g,
                          const float* __restrict__ b, __bf16* __restrict__ hff) {
  int warp = (blockIdx.x * blockDim.x + threadIdx.x) >> 5;
  int lane = threadIdx.x & 31;
  if (warp >= NN) return;
  const float* src = xmid + (long)warp*H;
  float v[4];
#pragma unroll
  for (int i = 0; i < 4; ++i) v[i] = src[lane + i*32];
  float s = v[0] + v[1] + v[2] + v[3];
#pragma unroll
  for (int msk = 1; msk < 32; msk <<= 1) s += __shfl_xor(s, msk, 32);
  float mu = s * (1.f/H);
  float vs = 0.f;
#pragma unroll
  for (int i = 0; i < 4; ++i) { float d = v[i] - mu; vs += d*d; }
#pragma unroll
  for (int msk = 1; msk < 32; msk <<= 1) vs += __shfl_xor(vs, msk, 32);
  float inv = rsqrtf(vs * (1.f/H) + 1e-5f);
#pragma unroll
  for (int i = 0; i < 4; ++i) {
    int c = lane + i*32;
    hff[(long)warp*H + c] = (__bf16)((v[i] - mu) * inv * g[c] + b[c]);
  }
}

// FFN 128->512->128 in 4 hidden chunks + residual + output permute.
// Slots 16..19 (ffw1 chunks), 20..23 (ffw2 chunks).
__global__ void k_ffn(const __bf16* __restrict__ hff, const float* __restrict__ xmid,
                      const __bf16* __restrict__ wsl,
                      const float* ffb1, const float* ffb2,
                      float* __restrict__ out) {
  __shared__ __attribute__((aligned(16))) __bf16 wcm[SLOT];
  __shared__ __attribute__((aligned(16))) __bf16 hbuf[4*16*H];
  int wave = threadIdx.x >> 5, lane = threadIdx.x & 31;
  int mrow = lane & 15, half = lane >> 4;
  long row0 = (long)blockIdx.x*64 + wave*16;
  __bf16* hb = hbuf + wave*16*H;

  v8f accO[8]; zero8(accO);
  for (int c = 0; c < 4; ++c) {
    __syncthreads();
    stage_async(wsl + (16 + c)*SLOT, wcm);
    __syncthreads();
    v8f acc1[8]; zero8(acc1);
    gemm_acc(hff + row0*H, H, 16, wcm, mrow, half, acc1);
#pragma unroll
    for (int j = 0; j < 8; ++j) {
      float bb = ffb1[c*H + j*16 + mrow];
#pragma unroll
      for (int r = 0; r < 8; ++r)
        hb[(r + half*8)*H + j*16 + mrow] = (__bf16)fmaxf(acc1[j][r] + bb, 0.f);
    }
    __syncthreads();
    stage_async(wsl + (20 + c)*SLOT, wcm);
    __syncthreads();
    gemm_acc(hb, H, 16, wcm, mrow, half, accO);
  }
#pragma unroll
  for (int j = 0; j < 8; ++j) {
    int col = j*16 + mrow; float bb = ffb2[col];
#pragma unroll
    for (int r = 0; r < 8; ++r) {
      long n = row0 + r + half*8;
      int a = (int)(n & 63), m = (int)((n >> 6) % MM), t = (int)(n / (AA*MM));
      out[((long)(a*MM + m)*TT + t)*H + col] = xmid[n*H + col] + accO[j][r] + bb;
    }
  }
}

// ---------------- host ----------------

extern "C" void kernel_launch(void* const* d_in, const int* in_sizes, int n_in,
                              void* d_out, int out_size, void* d_ws, size_t ws_size,
                              hipStream_t stream) {
  (void)in_sizes; (void)n_in; (void)out_size; (void)ws_size;
  const float* x       = (const float*)d_in[0];
  const float* pos     = (const float*)d_in[1];
  const float* head    = (const float*)d_in[2];
  const float* freqs   = (const float*)d_in[3];
  const float* mlp_w1  = (const float*)d_in[4];
  const float* mlp_b1  = (const float*)d_in[5];
  const float* mlp_ln_g= (const float*)d_in[6];
  const float* mlp_ln_b= (const float*)d_in[7];
  const float* mlp_w2  = (const float*)d_in[8];
  const float* mlp_b2  = (const float*)d_in[9];
  const float* out_ln_g= (const float*)d_in[10];
  const float* out_ln_b= (const float*)d_in[11];
  const float* out_w   = (const float*)d_in[12];
  const float* out_b   = (const float*)d_in[13];
  const float* pre_x_g = (const float*)d_in[14];
  const float* pre_x_b = (const float*)d_in[15];
  const float* pre_r_g = (const float*)d_in[16];
  const float* pre_r_b = (const float*)d_in[17];
  const float* wq = (const float*)d_in[18];
  const float* bq = (const float*)d_in[19];
  const float* wk = (const float*)d_in[20];
  const float* wv = (const float*)d_in[21];
  const float* bv = (const float*)d_in[22];
  const float* wkr= (const float*)d_in[23];
  const float* wvr= (const float*)d_in[24];
  const float* bvr= (const float*)d_in[25];
  const float* ws_= (const float*)d_in[26];
  const float* bs = (const float*)d_in[27];
  const float* wg = (const float*)d_in[28];
  const float* bg = (const float*)d_in[29];
  const float* wo = (const float*)d_in[30];
  const float* bo = (const float*)d_in[31];
  const float* ff_g = (const float*)d_in[32];
  const float* ff_b = (const float*)d_in[33];
  const float* ffw1 = (const float*)d_in[34];
  const float* ffb1 = (const float*)d_in[35];
  const float* ffw2 = (const float*)d_in[36];
  const float* ffb2 = (const float*)d_in[37];
  const int* edges  = (const int*)d_in[38];
  float* out = (float*)d_out;

  char* wptr = (char*)d_ws;
  auto alloc = [&](size_t bytes) -> char* {
    char* p = wptr; wptr += (bytes + 255) & ~(size_t)255; return p;
  };
  __bf16*   wsl   = (__bf16*)  alloc((size_t)24*SLOT*sizeof(__bf16));
  float*    rel   = (float*)   alloc((size_t)EE*3*sizeof(float));
  __bf16*   rnbf  = (__bf16*)  alloc((size_t)EE*H*sizeof(__bf16));
  float*    sim   = (float*)   alloc((size_t)EE*8*sizeof(float));
  __bf16*   vebf  = (__bf16*)  alloc((size_t)EE*H*sizeof(__bf16));
  float*    xf    = (float*)   alloc((size_t)NN*H*sizeof(float));
  __bf16*   xnbf  = (__bf16*)  alloc((size_t)NN*H*sizeof(__bf16));
  float*    qb    = (float*)   alloc((size_t)NN*H*sizeof(float));
  float*    kb    = (float*)   alloc((size_t)NN*H*sizeof(float));
  float*    vb    = (float*)   alloc((size_t)NN*H*sizeof(float));
  unsigned* smaxu = (unsigned*)alloc((size_t)NN*8*sizeof(unsigned));
  float*    den   = (float*)   alloc((size_t)NN*8*sizeof(float));
  float*    agg   = (float*)   alloc((size_t)NN*H*sizeof(float));
  float*    xmid  = (float*)   alloc((size_t)NN*H*sizeof(float));
  __bf16*   hff   = (__bf16*)  alloc((size_t)NN*H*sizeof(__bf16));

  hipMemsetAsync(smaxu, 0, (size_t)NN*8*sizeof(unsigned), stream);
  hipMemsetAsync(den,   0, (size_t)NN*8*sizeof(float), stream);
  hipMemsetAsync(agg,   0, (size_t)NN*H*sizeof(float), stream);

  // one-time weight image conversion (24 slots)
  const int CB = 256, CG = (SLOT + CB - 1)/CB;
  k_conv<<<CG, CB, 0, stream>>>(wq, H, H, wsl + 0*SLOT);
  k_conv<<<CG, CB, 0, stream>>>(wk, H, H, wsl + 1*SLOT);
  k_conv<<<CG, CB, 0, stream>>>(wv, H, H, wsl + 2*SLOT);
  for (int i = 0; i < 3; ++i) {
    k_conv<<<CG, CB, 0, stream>>>(mlp_w1 + (size_t)i*129*H, H, 129, wsl + (3+i)*SLOT);
    k_conv<<<CG, CB, 0, stream>>>(mlp_w2 + (size_t)i*H*H,   H, H,   wsl + (6+i)*SLOT);
  }
  k_conv<<<CG, CB, 0, stream>>>(out_w, H, H, wsl + 9*SLOT);
  k_conv<<<CG, CB, 0, stream>>>(wkr, H, H, wsl + 10*SLOT);
  k_conv<<<CG, CB, 0, stream>>>(wvr, H, H, wsl + 11*SLOT);
  k_conv<<<CG, CB, 0, stream>>>(wg, H, H, wsl + 12*SLOT);
  k_conv<<<CG, CB, 0, stream>>>(wg + (size_t)H*H, H, H, wsl + 13*SLOT);
  k_conv<<<CG, CB, 0, stream>>>(ws_, H, H, wsl + 14*SLOT);
  k_conv<<<CG, CB, 0, stream>>>(wo, H, H, wsl + 15*SLOT);
  for (int c = 0; c < 4; ++c) {
    k_conv<<<CG, CB, 0, stream>>>(ffw1 + (size_t)c*H, 4*H, H, wsl + (16+c)*SLOT);
    k_conv<<<CG, CB, 0, stream>>>(ffw2 + (size_t)c*H*H, H, H, wsl + (20+c)*SLOT);
  }

  k_edge_geom<<<(EE + 255)/256, 256, 0, stream>>>(edges, pos, head, rel);
  k_node_prep<<<(NN*32 + 255)/256, 256, 0, stream>>>(x, pre_x_g, pre_x_b, xf, xnbf);
  k_qkv<<<NN/64, 128, 0, stream>>>(xnbf, wsl, bq, bv, qb, kb, vb);
  k_fourier<<<(EE + 63)/64, 128, 0, stream>>>(rel, freqs, wsl, mlp_b1, mlp_ln_g, mlp_ln_b,
                                              mlp_b2, out_ln_g, out_ln_b, out_b,
                                              pre_r_g, pre_r_b, rnbf);
  k_attn<<<(EE + 63)/64, 128, 0, stream>>>(rnbf, edges, wsl, bvr, qb, kb, vb,
                                           sim, smaxu, vebf);
  k_soft2<<<(int)(((long)EE*8 + 255)/256), 256, 0, stream>>>(edges, smaxu, sim, den);
  k_agg<<<(int)(((long)EE*H + 255)/256), 256, 0, stream>>>(edges, sim, den, vebf, agg);
  k_update<<<NN/64, 128, 0, stream>>>(xnbf, xf, agg, wsl, bg, bs, bo, xmid);
  k_ffnprep<<<(NN*32 + 255)/256, 256, 0, stream>>>(xmid, ff_g, ff_b, hff);
  k_ffn<<<NN/64, 128, 0, stream>>>(hff, xmid, wsl, ffb1, ffb2, out);
}
